// PMA_35424890257444
// MI455X (gfx1250) — matrix operand order repaired and checked by reading, not compiled, not run on hardware
//
#include <hip/hip_runtime.h>
#include <hip/hip_bf16.h>

// ---------------------------------------------------------------------------
// PMA (Set-Transformer pooling) for MI455X / gfx1250.
// B=8, L=4096, D=1024, H=16, NS=32, hd=64.
// Dominant work: K/V projections (137 GFLOP) -> bf16 WMMA GEMM.
// All WMMA fragments are laid out so each lane's 16 bf16 elements are a
// contiguous 32B run -> 2x b128 loads (no scalar ds_load_u16 chains).
// ---------------------------------------------------------------------------

#define B_   8
#define L_   4096
#define D_   1024
#define H_   16
#define NS_  32
#define HD_  64
#define EPS_ 1e-5f
#define SCALE_ 0.03125f   // 1/sqrt(D)

typedef __attribute__((ext_vector_type(16))) __bf16 bf16x16;
typedef __attribute__((ext_vector_type(8)))  float  f32x8;

union FragU { bf16x16 v; uint4 u[2]; };

// ======================= 1. Q projection (tiny) ============================
__global__ __launch_bounds__(256) void qproj_kernel(
    const float* __restrict__ S, const float* __restrict__ Wq,
    const float* __restrict__ bq, float* __restrict__ Qp) {
  int idx = blockIdx.x * 256 + threadIdx.x;   // 0..32767
  int q = idx >> 10, o = idx & (D_ - 1);
  const float4* s = (const float4*)(S + (size_t)q * D_);
  const float4* w = (const float4*)(Wq + (size_t)o * D_);
  float acc = 0.f;
#pragma unroll 8
  for (int i = 0; i < D_ / 4; ++i) {
    float4 a = s[i], b = w[i];
    acc += a.x * b.x + a.y * b.y + a.z * b.z + a.w * b.w;
  }
  Qp[idx] = acc + bq[o];
}

// ======================= 2. K/V projection GEMM ============================
// C[l, o] = sum_d X[b,l,d] * W[o,d] + bias[o], o in [0,2048): [Wk | Wv].
// K stored [B,H,L,64] (row-major); V stored transposed [B,H,64,L].
// Block tile 128x128, K-step 64 (2 WMMA sub-steps), 8 waves of 32x64 tiles.
__global__ __launch_bounds__(256) void kv_proj_kernel(
    const float* __restrict__ X,
    const float* __restrict__ Wk, const float* __restrict__ bk,
    const float* __restrict__ Wv, const float* __restrict__ bv,
    __bf16* __restrict__ Kb, __bf16* __restrict__ Vb) {
  __shared__ __bf16 As[128][72];   // [l][k], row pad 72 (144B, 16B aligned)
  __shared__ __bf16 Bs[128][72];   // [o][k] -- transposed by construction
  const int tid = threadIdx.x, lane = tid & 31, wave = tid >> 5;
  const int half = lane >> 4;
  const int b = blockIdx.z, l0 = blockIdx.y * 128, o0 = blockIdx.x * 128;
  const float* W = (o0 < D_) ? Wk : Wv;
  const int oW = o0 & (D_ - 1);
  const int m0w = (wave >> 1) * 32;   // wave's M offset (0..96)
  const int n0w = (wave & 1) * 64;    // wave's N offset (0 or 64)

  f32x8 zero8 = {0.f, 0.f, 0.f, 0.f, 0.f, 0.f, 0.f, 0.f};
  f32x8 acc[2][4];
#pragma unroll
  for (int mt = 0; mt < 2; ++mt)
#pragma unroll
    for (int nt = 0; nt < 4; ++nt) acc[mt][nt] = zero8;

  const int r2 = tid >> 1, seg = tid & 1;   // staging: 2 threads per row
  for (int kk = 0; kk < D_; kk += 64) {
    __syncthreads();
    { // stage A: X[l0+r2][kk+seg*32 .. +31] -> As[r2][seg*32..]
      const float4* src =
          (const float4*)(X + (size_t)(b * L_ + l0 + r2) * D_ + kk + seg * 32);
      float4 t[8];
#pragma unroll
      for (int j = 0; j < 8; ++j) t[j] = src[j];
      const float* f = (const float*)t;
      __bf16* dst = &As[r2][seg * 32];
#pragma unroll
      for (int j = 0; j < 32; ++j) dst[j] = (__bf16)f[j];
      if (kk + 64 < D_)  // pull next A tile toward L2 (global_prefetch_b8)
        __builtin_prefetch(X + (size_t)(b * L_ + l0 + r2) * D_ + kk + 64, 0, 1);
    }
    { // stage B: W[oW+r2][kk+seg*32 .. +31] -> Bs[r2][seg*32..]  (no transpose)
      const float4* src =
          (const float4*)(W + (size_t)(oW + r2) * D_ + kk + seg * 32);
      float4 t[8];
#pragma unroll
      for (int j = 0; j < 8; ++j) t[j] = src[j];
      const float* f = (const float*)t;
      __bf16* dst = &Bs[r2][seg * 32];
#pragma unroll
      for (int j = 0; j < 32; ++j) dst[j] = (__bf16)f[j];
    }
    __syncthreads();

#pragma unroll
    for (int ks = 0; ks < 2; ++ks) {
      // A frags: lane row = M, elements contiguous along k (2 x 16B runs)
      FragU a[2], bb[4];
#pragma unroll
      for (int mt = 0; mt < 2; ++mt) {
        int mrow = m0w + mt * 16 + (lane & 15);
        a[mt].u[0] = *(const uint4*)&As[mrow][ks * 32 + half * 16];
        a[mt].u[1] = *(const uint4*)&As[mrow][ks * 32 + half * 16 + 8];
      }
      // B frags: lane col = N, elements contiguous along k (1 x 32B run)
#pragma unroll
      for (int nt = 0; nt < 4; ++nt) {
        int ncol = n0w + nt * 16 + (lane & 15);
        bb[nt].u[0] = *(const uint4*)&Bs[ncol][ks * 32 + half * 16];
        bb[nt].u[1] = *(const uint4*)&Bs[ncol][ks * 32 + half * 16 + 8];
      }
#pragma unroll
      for (int mt = 0; mt < 2; ++mt)
#pragma unroll
        for (int nt = 0; nt < 4; ++nt)
          acc[mt][nt] = __builtin_amdgcn_wmma_f32_16x16x32_bf16(
              false, a[mt].v, false, bb[nt].v, (short)0, acc[mt][nt],
              false, false);
    }
  }

  // epilogue: bias, bf16 convert, route to K ([l][e]) or V (transposed [e][l])
#pragma unroll
  for (int mt = 0; mt < 2; ++mt)
#pragma unroll
    for (int nt = 0; nt < 4; ++nt)
#pragma unroll
      for (int r = 0; r < 8; ++r) {
        int l = l0 + m0w + mt * 16 + r + half * 8;
        int o = o0 + n0w + nt * 16 + (lane & 15);
        float val = acc[mt][nt][r];
        if (o < D_) {
          val += bk[o];
          int h = o >> 6, e = o & 63;
          Kb[((size_t)(b * H_ + h) * L_ + l) * HD_ + e] = (__bf16)val;
        } else {
          int ov = o - D_;
          val += bv[ov];
          int h = ov >> 6, e = ov & 63;
          Vb[(((size_t)(b * H_ + h)) * HD_ + e) * L_ + l] = (__bf16)val;
        }
      }
}

// ======================= 3. Flash attention (per b,h) ======================
// 4 waves/block, each owns 1024 keys (32-key tiles), online softmax.
// K/V fragments load straight from global as contiguous 32B runs:
//   QK^T B-frag: K[key][e0..e0+15]   (K row-major)
//   P.V  B-frag: V[e][key0..key0+15] (V transposed)
__global__ __launch_bounds__(128) void attn_kernel(
    const float* __restrict__ Qp,      // [NS, D]
    const __bf16* __restrict__ Kb,     // [B*H, L, 64]
    const __bf16* __restrict__ Vb,     // [B*H, 64, L]
    const int* __restrict__ pad,       // [B, L]
    float* __restrict__ Opart,         // [B*H, 4, NS, 64]
    float* __restrict__ mw,            // [B*H, 4, NS]
    float* __restrict__ sw) {          // [B*H, 4, NS]
  __shared__ __align__(16) __bf16 Qs[NS_][HD_];     // 4 KB
  __shared__ float  Sc[4][NS_][33];                 // 16.5 KB (pad: row reads)
  __shared__ __align__(16) __bf16 Ps[4][NS_][40];   // 10 KB (80B rows, 16B ok)
  __shared__ float  resc[4][NS_];
  const int tid = threadIdx.x, lane = tid & 31, w = tid >> 5;
  const int half = lane >> 4;
  const int bh = blockIdx.x, b = bh >> 4, h = bh & 15;

  for (int i = tid; i < NS_ * HD_; i += 128) {
    int q = i >> 6, e = i & 63;
    Qs[q][e] = (__bf16)Qp[q * D_ + h * HD_ + e];
  }
  __syncthreads();

  // Q A-fragments: [mtile][kstep over e], contiguous 16B runs
  FragU aq[2][2];
#pragma unroll
  for (int mt = 0; mt < 2; ++mt)
#pragma unroll
    for (int ks = 0; ks < 2; ++ks) {
      int qrow = mt * 16 + (lane & 15);
      aq[mt][ks].u[0] = *(const uint4*)&Qs[qrow][ks * 32 + half * 16];
      aq[mt][ks].u[1] = *(const uint4*)&Qs[qrow][ks * 32 + half * 16 + 8];
    }

  f32x8 zero8 = {0.f, 0.f, 0.f, 0.f, 0.f, 0.f, 0.f, 0.f};
  f32x8 accO[2][4];
#pragma unroll
  for (int mt = 0; mt < 2; ++mt)
#pragma unroll
    for (int nt = 0; nt < 4; ++nt) accO[mt][nt] = zero8;
  float m = -3.0e38f, s = 0.f;   // per-lane stats: this lane owns q == lane

  const __bf16* Kbh = Kb + (size_t)bh * L_ * HD_;
  const __bf16* Vbh = Vb + (size_t)bh * HD_ * L_;

  for (int it = 0; it < 32; ++it) {
    int key0 = w * 1024 + it * 32;

    // scores = Q @ K^T  (M=32 q, N=32 keys, K=64 e -> 8 WMMA)
    f32x8 accS[2][2];
    accS[0][0] = zero8; accS[0][1] = zero8;
    accS[1][0] = zero8; accS[1][1] = zero8;
#pragma unroll
    for (int ks = 0; ks < 2; ++ks) {
      FragU bkf[2];
#pragma unroll
      for (int nt = 0; nt < 2; ++nt) {
        int key = key0 + nt * 16 + (lane & 15);
        const __bf16* p = Kbh + (size_t)key * HD_ + ks * 32 + half * 16;
        bkf[nt].u[0] = *(const uint4*)p;
        bkf[nt].u[1] = *(const uint4*)(p + 8);
      }
#pragma unroll
      for (int mt = 0; mt < 2; ++mt)
#pragma unroll
        for (int nt = 0; nt < 2; ++nt)
          accS[mt][nt] = __builtin_amdgcn_wmma_f32_16x16x32_bf16(
              false, aq[mt][ks].v, false, bkf[nt].v, (short)0, accS[mt][nt],
              false, false);
    }

    // dump masked scores to LDS
#pragma unroll
    for (int nt = 0; nt < 2; ++nt) {
      int key = nt * 16 + (lane & 15);
      int pm = pad[b * L_ + key0 + key];
#pragma unroll
      for (int mt = 0; mt < 2; ++mt)
#pragma unroll
        for (int r = 0; r < 8; ++r) {
          int q = mt * 16 + r + half * 8;
          float v = accS[mt][nt][r] * SCALE_;
          Sc[w][q][key] = pm ? v : -1.0e30f;
        }
    }

    // online softmax: lane owns row q = lane
    float tmax = -3.0e38f;
#pragma unroll
    for (int j = 0; j < 32; ++j) tmax = fmaxf(tmax, Sc[w][lane][j]);
    float mnew = fmaxf(m, tmax);
    float scaleOld = __expf(m - mnew);
    float ssum = 0.f;
#pragma unroll
    for (int j = 0; j < 32; ++j) {
      float p = __expf(Sc[w][lane][j] - mnew);
      ssum += p;
      Ps[w][lane][j] = (__bf16)p;
    }
    s = s * scaleOld + ssum;
    m = mnew;
    resc[w][lane] = scaleOld;

    // rescale running O, then O += P @ V  (K=32 keys -> 8 WMMA)
#pragma unroll
    for (int mt = 0; mt < 2; ++mt)
#pragma unroll
      for (int nt = 0; nt < 4; ++nt)
#pragma unroll
        for (int r = 0; r < 8; ++r) {
          int q = mt * 16 + r + half * 8;
          accO[mt][nt][r] *= resc[w][q];
        }
    FragU ap[2];
#pragma unroll
    for (int mt = 0; mt < 2; ++mt) {
      int qrow = mt * 16 + (lane & 15);
      ap[mt].u[0] = *(const uint4*)&Ps[w][qrow][half * 16];
      ap[mt].u[1] = *(const uint4*)&Ps[w][qrow][half * 16 + 8];
    }
#pragma unroll
    for (int nt = 0; nt < 4; ++nt) {
      int e = nt * 16 + (lane & 15);
      const __bf16* p = Vbh + (size_t)e * L_ + key0 + half * 16;
      FragU bvf;
      bvf.u[0] = *(const uint4*)p;
      bvf.u[1] = *(const uint4*)(p + 8);
#pragma unroll
      for (int mt = 0; mt < 2; ++mt)
        accO[mt][nt] = __builtin_amdgcn_wmma_f32_16x16x32_bf16(
            false, ap[mt].v, false, bvf.v, (short)0, accO[mt][nt],
            false, false);
    }
  }

  // write wave partials
  float* Op = Opart + ((size_t)bh * 4 + w) * NS_ * HD_;
#pragma unroll
  for (int mt = 0; mt < 2; ++mt)
#pragma unroll
    for (int nt = 0; nt < 4; ++nt)
#pragma unroll
      for (int r = 0; r < 8; ++r) {
        int q = mt * 16 + r + half * 8;
        int e = nt * 16 + (lane & 15);
        Op[q * HD_ + e] = accO[mt][nt][r];
      }
  mw[((size_t)bh * 4 + w) * NS_ + lane] = m;
  sw[((size_t)bh * 4 + w) * NS_ + lane] = s;
}

// ======================= 4. combine wave partials ==========================
__global__ __launch_bounds__(256) void attn_reduce_kernel(
    const float* __restrict__ Opart, const float* __restrict__ mw,
    const float* __restrict__ sw, float* __restrict__ Oacc) {
  int bh = blockIdx.x, b = bh >> 4, h = bh & 15;
  for (int i = threadIdx.x; i < NS_ * HD_; i += 256) {
    int q = i >> 6, e = i & 63;
    float M = -3.0e38f;
#pragma unroll
    for (int w = 0; w < 4; ++w) M = fmaxf(M, mw[(bh * 4 + w) * NS_ + q]);
    float den = 0.f, num = 0.f;
#pragma unroll
    for (int w = 0; w < 4; ++w) {
      float f = __expf(mw[(bh * 4 + w) * NS_ + q] - M);
      den += sw[(bh * 4 + w) * NS_ + q] * f;
      num += Opart[((size_t)(bh * 4 + w) * NS_ + q) * HD_ + e] * f;
    }
    Oacc[((size_t)(b * NS_) + q) * D_ + h * HD_ + e] =
        (den > 0.f) ? num / den : 0.f;
  }
}

// ======================= 5. residual + LayerNorm 0 =========================
__global__ __launch_bounds__(256) void ln0_kernel(
    const float* __restrict__ S, const float* __restrict__ Oacc,
    const float* __restrict__ g0, const float* __restrict__ b0,
    float* __restrict__ R) {
  __shared__ float red[256];
  int row = blockIdx.x;     // b*32 + q
  int q = row & 31, t = threadIdx.x;
  float x[4]; float sum = 0.f;
#pragma unroll
  for (int j = 0; j < 4; ++j) {
    int d = t + 256 * j;
    x[j] = S[q * D_ + d] + Oacc[(size_t)row * D_ + d];  // residual uses Q = S
    sum += x[j];
  }
  red[t] = sum; __syncthreads();
  for (int st = 128; st > 0; st >>= 1) {
    if (t < st) red[t] += red[t + st];
    __syncthreads();
  }
  float mu = red[0] / D_;
  __syncthreads();
  float vs = 0.f;
#pragma unroll
  for (int j = 0; j < 4; ++j) { float d2 = x[j] - mu; vs += d2 * d2; }
  red[t] = vs; __syncthreads();
  for (int st = 128; st > 0; st >>= 1) {
    if (t < st) red[t] += red[t + st];
    __syncthreads();
  }
  float inv = rsqrtf(red[0] / D_ + EPS_);
#pragma unroll
  for (int j = 0; j < 4; ++j) {
    int d = t + 256 * j;
    R[(size_t)row * D_ + d] = (x[j] - mu) * inv * g0[d] + b0[d];
  }
}

// ======================= 6. MLP + residual + LayerNorm 1 ===================
__global__ __launch_bounds__(256) void mlp_ln1_kernel(
    const float* __restrict__ R, const float* __restrict__ Wo,
    const float* __restrict__ bo, const float* __restrict__ g1,
    const float* __restrict__ b1, float* __restrict__ out) {
  __shared__ float row[D_];
  __shared__ float red[256];
  int rix = blockIdx.x, t = threadIdx.x;
#pragma unroll
  for (int j = 0; j < 4; ++j)
    row[t + 256 * j] = R[(size_t)rix * D_ + t + 256 * j];
  __syncthreads();
  float u[4]; float sum = 0.f;
#pragma unroll
  for (int j = 0; j < 4; ++j) {
    int o = t + 256 * j;
    const float4* wv = (const float4*)(Wo + (size_t)o * D_);
    float acc = 0.f;
#pragma unroll 4
    for (int i = 0; i < D_ / 4; ++i) {
      float4 wq = wv[i];
      acc += row[4 * i] * wq.x + row[4 * i + 1] * wq.y +
             row[4 * i + 2] * wq.z + row[4 * i + 3] * wq.w;
    }
    acc += bo[o];
    u[j] = row[o] + fmaxf(acc, 0.f);
    sum += u[j];
  }
  red[t] = sum; __syncthreads();
  for (int st = 128; st > 0; st >>= 1) {
    if (t < st) red[t] += red[t + st];
    __syncthreads();
  }
  float mu = red[0] / D_;
  __syncthreads();
  float vs = 0.f;
#pragma unroll
  for (int j = 0; j < 4; ++j) { float d2 = u[j] - mu; vs += d2 * d2; }
  red[t] = vs; __syncthreads();
  for (int st = 128; st > 0; st >>= 1) {
    if (t < st) red[t] += red[t + st];
    __syncthreads();
  }
  float inv = rsqrtf(red[0] / D_ + EPS_);
#pragma unroll
  for (int j = 0; j < 4; ++j) {
    int o = t + 256 * j;
    out[(size_t)rix * D_ + o] = (u[j] - mu) * inv * g1[o] + b1[o];
  }
}

// ======================= launch ============================================
extern "C" void kernel_launch(void* const* d_in, const int* in_sizes, int n_in,
                              void* d_out, int out_size, void* d_ws,
                              size_t ws_size, hipStream_t stream) {
  const float* X   = (const float*)d_in[0];
  const int*   pad = (const int*)d_in[1];
  const float* S   = (const float*)d_in[2];
  const float* Wq  = (const float*)d_in[3];
  const float* bq  = (const float*)d_in[4];
  const float* Wk  = (const float*)d_in[5];
  const float* bk  = (const float*)d_in[6];
  const float* Wv  = (const float*)d_in[7];
  const float* bv  = (const float*)d_in[8];
  const float* Wo  = (const float*)d_in[9];
  const float* bo  = (const float*)d_in[10];
  const float* g0  = (const float*)d_in[11];
  const float* b0  = (const float*)d_in[12];
  const float* g1  = (const float*)d_in[13];
  const float* b1  = (const float*)d_in[14];

  char* ws = (char*)d_ws;
  size_t off = 0;
  __bf16* Kb    = (__bf16*)(ws + off); off += (size_t)B_ * H_ * L_ * HD_ * 2;
  __bf16* Vb    = (__bf16*)(ws + off); off += (size_t)B_ * H_ * HD_ * L_ * 2;
  float*  Qp    = (float*)(ws + off);  off += (size_t)NS_ * D_ * 4;
  float*  Opart = (float*)(ws + off);  off += (size_t)B_ * H_ * 4 * NS_ * HD_ * 4;
  float*  mw    = (float*)(ws + off);  off += (size_t)B_ * H_ * 4 * NS_ * 4;
  float*  sw    = (float*)(ws + off);  off += (size_t)B_ * H_ * 4 * NS_ * 4;
  float*  Oacc  = (float*)(ws + off);  off += (size_t)B_ * NS_ * D_ * 4;
  float*  R     = (float*)(ws + off);  off += (size_t)B_ * NS_ * D_ * 4;

  qproj_kernel<<<(NS_ * D_) / 256, 256, 0, stream>>>(S, Wq, bq, Qp);
  kv_proj_kernel<<<dim3(2 * D_ / 128, L_ / 128, B_), 256, 0, stream>>>(
      X, Wk, bk, Wv, bv, Kb, Vb);
  attn_kernel<<<B_ * H_, 128, 0, stream>>>(Qp, Kb, Vb, pad, Opart, mw, sw);
  attn_reduce_kernel<<<B_ * H_, 256, 0, stream>>>(Opart, mw, sw, Oacc);
  ln0_kernel<<<B_ * NS_, 256, 0, stream>>>(S, Oacc, g0, b0, R);
  mlp_ln1_kernel<<<B_ * NS_, 256, 0, stream>>>(R, Wo, bo, g1, b1,
                                               (float*)d_out);
}
